// Attention_61830349193786
// MI455X (gfx1250) — compile-verified, matrix operand order checked by reading, not run on hardware
//
#include <hip/hip_runtime.h>
#include <hip/hip_bf16.h>
#include <cstdint>

typedef __attribute__((ext_vector_type(16))) _Float16 v16h;
typedef __attribute__((ext_vector_type(8)))  float    v8f;
typedef __attribute__((ext_vector_type(4)))  unsigned int u32x4;
typedef __attribute__((ext_vector_type(8)))  int          i32x8;
typedef __attribute__((ext_vector_type(4)))  int          i32x4;

__device__ __forceinline__ v8f wmma_f16(v16h a, v16h b, v8f c) {
    return __builtin_amdgcn_wmma_f32_16x16x32_f16(
        false, a, false, b, (short)0, c, false, false);
}

// v_cvt_pk_rtz_f16_f32: pack two f32 -> two f16 in one u32
__device__ __forceinline__ unsigned int pk2(float a, float b) {
    auto h = __builtin_amdgcn_cvt_pkrtz(a, b);
    return __builtin_bit_cast(unsigned int, h);
}

// ---------------------------------------------------------------------------
// Tensor Data Mover: 2D tile load Global -> LDS.
// Builds D# groups 0/1 per CDNA5 ISA ch.8 (groups 2/3 zero => <=2D tensor).
//   dsz: 0=1B 1=2B 2=4B 3=8B; tile0/tile1 in elements; stride0 in elements.
// Issue from ONE wave; follow with s_wait_tensorcnt(0) + workgroup barrier.
// ---------------------------------------------------------------------------
__device__ __forceinline__ void tdm_load_2d(const void* gptr, unsigned int lds_off,
                                            unsigned int dsz, unsigned int tile0,
                                            unsigned int tile1,
                                            unsigned long long stride0)
{
    unsigned long long ga = (unsigned long long)(uintptr_t)gptr;
    u32x4 g0 = { 1u,                                   // count=1 (valid user D#)
                 lds_off,                              // lds_addr (bytes)
                 (unsigned int)(ga & 0xffffffffull),   // global_addr[31:0]
                 (unsigned int)((ga >> 32) & 0x1ffffffull) | (2u << 30) }; // [56:32]|type=2
    i32x8 g1 = {
        (int)(dsz << 16),                                              // mask=0,data_size
        (int)((tile0 & 0xffffu) << 16),                                // tensor_dim0 lo
        (int)(((tile0 >> 16) & 0xffffu) | ((tile1 & 0xffffu) << 16)),  // td0 hi | td1 lo
        (int)(((tile1 >> 16) & 0xffffu) | ((tile0 & 0xffffu) << 16)),  // td1 hi | tile_dim0
        (int)(tile1 & 0xffffu),                                        // tile_dim1 (dim2=0)
        (int)(stride0 & 0xffffffffull),                                // dim0_stride lo
        (int)((stride0 >> 32) & 0xffffull),                            // dim0_stride hi
        0 };
    i32x4 gz = { 0, 0, 0, 0 };
#if __clang_major__ >= 23
    i32x8 gz8 = { 0, 0, 0, 0, 0, 0, 0, 0 };
    __builtin_amdgcn_tensor_load_to_lds(g0, g1, gz, gz, gz8, 0);
#else
    __builtin_amdgcn_tensor_load_to_lds(g0, g1, gz, gz, 0);
#endif
}

#define PK 1024
#define PN 1024

// ---------------------------------------------------------------------------
// QKV projection: out = split_heads(X @ W + bias) * scale
// 128x128 block tile, 8 waves (2x4 of 64x32), BK=32, f32->f16 via cvt_pkrtz.
// transposed=0 -> out[bh][n][d] (Q/K);  transposed=1 -> out[bh][d][n] (V^T).
// ---------------------------------------------------------------------------
__global__ __launch_bounds__(256)
void qkv_proj_kernel(const float* __restrict__ X, const float* __restrict__ W,
                     const float* __restrict__ bias, _Float16* __restrict__ out,
                     int transposed, float scale)
{
    __shared__ _Float16 Ash[128 * 32];   // [m][k]
    __shared__ _Float16 Bsh[128 * 32];   // [n][k]

    const int tid  = threadIdx.x;
    const int lane = tid & 31;
    const int wave = tid >> 5;
    const int wr   = wave >> 2;
    const int wc   = wave & 3;
    const int m0   = blockIdx.x * 128;
    const int n0   = blockIdx.y * 128;
    const int lh   = lane >> 4;
    const int ln   = lane & 15;

    v8f acc[4][2];
#pragma unroll
    for (int i = 0; i < 4; ++i)
#pragma unroll
        for (int j = 0; j < 2; ++j) acc[i][j] = (v8f){0,0,0,0,0,0,0,0};

    for (int k0 = 0; k0 < PK; k0 += 32) {
        // A tile: 2 threads/row, 16 floats each -> 8 packed u32 -> 2x b128 DS stores
        {
            const int r  = tid >> 1;
            const int kk = (tid & 1) * 16;
            const float4* src = (const float4*)&X[(size_t)(m0 + r) * PK + k0 + kk];
            unsigned int p[8];
#pragma unroll
            for (int i = 0; i < 4; ++i) {
                float4 f = src[i];
                p[2*i+0] = pk2(f.x, f.y);
                p[2*i+1] = pk2(f.z, f.w);
            }
            uint4* dst = (uint4*)&Ash[r * 32 + kk];
            dst[0] = make_uint4(p[0], p[1], p[2], p[3]);
            dst[1] = make_uint4(p[4], p[5], p[6], p[7]);
        }
        // W tile: each thread covers 2 k-rows x 8 cols; packed k-pairs -> b32 stores
        {
            const int nn = (tid & 15) * 8;        // 0..120
            const int kp = tid >> 4;              // 0..15
            const int k  = 2 * kp;
            const float4* r0 = (const float4*)&W[(size_t)(k0 + k)     * PN + n0 + nn];
            const float4* r1 = (const float4*)&W[(size_t)(k0 + k + 1) * PN + n0 + nn];
            float a[8], b[8];
            float4 t;
            t = r0[0]; a[0]=t.x; a[1]=t.y; a[2]=t.z; a[3]=t.w;
            t = r0[1]; a[4]=t.x; a[5]=t.y; a[6]=t.z; a[7]=t.w;
            t = r1[0]; b[0]=t.x; b[1]=t.y; b[2]=t.z; b[3]=t.w;
            t = r1[1]; b[4]=t.x; b[5]=t.y; b[6]=t.z; b[7]=t.w;
#pragma unroll
            for (int j = 0; j < 8; ++j)
                *(unsigned int*)&Bsh[(nn + j) * 32 + k] = pk2(a[j], b[j]);
        }
        __syncthreads();

        const int kb = lh * 16;
        v16h bfrag[2];
#pragma unroll
        for (int nt = 0; nt < 2; ++nt)
            bfrag[nt] = *(const v16h*)&Bsh[(wc*32 + nt*16 + ln) * 32 + kb];
#pragma unroll
        for (int mt = 0; mt < 4; ++mt) {
            v16h a = *(const v16h*)&Ash[(wr*64 + mt*16 + ln) * 32 + kb];
#pragma unroll
            for (int nt = 0; nt < 2; ++nt)
                acc[mt][nt] = wmma_f16(a, bfrag[nt], acc[mt][nt]);
        }
        __syncthreads();
    }

    if (transposed) {
        // V^T: for fixed (mt,nt) the 8 rows are consecutive n -> one b128 store
#pragma unroll
        for (int mt = 0; mt < 4; ++mt)
#pragma unroll
            for (int nt = 0; nt < 2; ++nt) {
                const int gm0 = m0 + wr*64 + mt*16 + 8*lh;      // n base (mult of 8)
                const int gn  = n0 + wc*32 + nt*16 + ln;
                const int b = gm0 >> 10, nb = gm0 & 1023;
                const int d = gn >> 4,  h = gn & 15;
                const float bb = bias[gn];
                float vv[8];
#pragma unroll
                for (int r = 0; r < 8; ++r) vv[r] = (acc[mt][nt][r] + bb) * scale;
                uint4 u = make_uint4(pk2(vv[0], vv[1]), pk2(vv[2], vv[3]),
                                     pk2(vv[4], vv[5]), pk2(vv[6], vv[7]));
                *(uint4*)&out[((size_t)(b*16 + h) * 64 + d) * 1024 + nb] = u;
            }
    } else {
#pragma unroll
        for (int mt = 0; mt < 4; ++mt)
#pragma unroll
            for (int nt = 0; nt < 2; ++nt)
#pragma unroll
                for (int r = 0; r < 8; ++r) {
                    const int gm = m0 + wr*64 + mt*16 + r + 8*lh;
                    const int gn = n0 + wc*32 + nt*16 + ln;
                    const float vv = (acc[mt][nt][r] + bias[gn]) * scale;
                    const int b = gm >> 10, n = gm & 1023;
                    const int d = gn >> 4,  h = gn & 15;
                    out[((size_t)(b*16 + h) * 1024 + n) * 64 + d] = (_Float16)vv;
                }
    }
}

// ---------------------------------------------------------------------------
// Flash attention with TDM-staged K/V/mask tiles.
// Block = (b,h) x 64 Q rows; 4 waves; wave 0 drives the Tensor Data Mover.
// ---------------------------------------------------------------------------
__global__ __launch_bounds__(128)
void attn_kernel(const _Float16* __restrict__ Qh, const _Float16* __restrict__ Kh,
                 const _Float16* __restrict__ Vt, const float* __restrict__ mask,
                 _Float16* __restrict__ O)
{
    __shared__ float    Msh[64 * 64];    // mask tile  (16 KB)
    __shared__ _Float16 Ksh[64 * 64];    // K chunk    ( 8 KB)  [n][d]
    __shared__ _Float16 Vsh[64 * 64];    // V chunk    ( 8 KB)  [d][n]
    __shared__ _Float16 Psh[64 * 64];    // P tile     ( 8 KB)

    const int tid  = threadIdx.x;
    const int lane = tid & 31;
    const int wave = tid >> 5;
    const int lh   = lane >> 4;
    const int ln   = lane & 15;
    const int bh   = blockIdx.y;
    const int q0   = blockIdx.x * 64;

    const _Float16* Qbase = Qh + ((size_t)bh * 1024 + q0) * 64;
    const _Float16* Kbase = Kh + (size_t)bh * 1024 * 64;
    const _Float16* Vbase = Vt + (size_t)bh * 64 * 1024;

    const int qrow = wave*16 + ln;
    v16h qf[2];
    qf[0] = *(const v16h*)&Qbase[qrow*64 +  0 + lh*16];
    qf[1] = *(const v16h*)&Qbase[qrow*64 + 32 + lh*16];

    v8f oacc[4];
#pragma unroll
    for (int dt = 0; dt < 4; ++dt) oacc[dt] = (v8f){0,0,0,0,0,0,0,0};
    float mrun[8], lrun[8];
#pragma unroll
    for (int r = 0; r < 8; ++r) { mrun[r] = -1.0e30f; lrun[r] = 0.0f; }

    const unsigned int koff = (unsigned int)(uintptr_t)&Ksh[0];
    const unsigned int voff = (unsigned int)(uintptr_t)&Vsh[0];
    const unsigned int moff = (unsigned int)(uintptr_t)&Msh[0];

    for (int kc = 0; kc < 1024; kc += 64) {
        if (wave == 0) {   // TDM issue (EXEC-independent, per-wave TENSORcnt)
            // K chunk: 64 rows x 64 d, contiguous -> 1 row of 4096 f16
            tdm_load_2d(&Kbase[(size_t)kc * 64], koff, 1u, 4096u, 1u, 4096ull);
            // V chunk: 64 d-rows x 64 n, row stride 1024 f16
            tdm_load_2d(&Vbase[kc], voff, 1u, 64u, 64u, 1024ull);
            // mask tile: 64 q-rows x 64 k, row stride 1024 f32
            tdm_load_2d(&mask[(size_t)q0 * 1024 + kc], moff, 2u, 64u, 64u, 1024ull);
            __builtin_amdgcn_s_wait_tensorcnt(0);
        }
        __syncthreads();

        // S = Q @ K^T for this wave's 16 rows x 64 chunk cols
        v8f sacc[4];
#pragma unroll
        for (int ct = 0; ct < 4; ++ct) {
            const int kr = ct*16 + ln;
            v16h b0 = *(const v16h*)&Ksh[kr*64 +  0 + lh*16];
            v16h b1 = *(const v16h*)&Ksh[kr*64 + 32 + lh*16];
            v8f s = (v8f){0,0,0,0,0,0,0,0};
            s = wmma_f16(qf[0], b0, s);
            s = wmma_f16(qf[1], b1, s);
            sacc[ct] = s;
        }

        // Online softmax
#pragma unroll
        for (int r = 0; r < 8; ++r) {
            const int lrow = wave*16 + r + 8*lh;
            float s[4];
#pragma unroll
            for (int ct = 0; ct < 4; ++ct)
                s[ct] = sacc[ct][r]
                      - 100000000.0f * (1.0f - Msh[lrow*64 + ct*16 + ln]);
            float mx = fmaxf(fmaxf(s[0], s[1]), fmaxf(s[2], s[3]));
#pragma unroll
            for (int off = 8; off > 0; off >>= 1)
                mx = fmaxf(mx, __shfl_xor(mx, off, 32));
            const float mnew  = fmaxf(mrun[r], mx);
            const float alpha = __expf(mrun[r] - mnew);
            float psum = 0.0f;
#pragma unroll
            for (int ct = 0; ct < 4; ++ct) {
                float p = __expf(s[ct] - mnew);
                Psh[lrow*64 + ct*16 + ln] = (_Float16)p;
                psum += p;
            }
#pragma unroll
            for (int off = 8; off > 0; off >>= 1)
                psum += __shfl_xor(psum, off, 32);
            lrun[r] = lrun[r] * alpha + psum;
            mrun[r] = mnew;
#pragma unroll
            for (int dt = 0; dt < 4; ++dt) oacc[dt][r] *= alpha;
        }
        __syncthreads();

        // O += P @ V
#pragma unroll
        for (int dt = 0; dt < 4; ++dt) {
            const int d = dt*16 + ln;
#pragma unroll
            for (int ks = 0; ks < 2; ++ks) {
                v16h a  = *(const v16h*)&Psh[(wave*16 + ln)*64 + ks*32 + lh*16];
                v16h bf = *(const v16h*)&Vsh[d*64 + ks*32 + lh*16];
                oacc[dt] = wmma_f16(a, bf, oacc[dt]);
            }
        }
        __syncthreads();   // protect Ksh/Vsh/Msh/Psh before next chunk's TDM
    }

    const int bb = bh >> 4, hh = bh & 15;
#pragma unroll
    for (int r = 0; r < 8; ++r) {
        const float inv = 1.0f / lrun[r];
        const int qn = q0 + wave*16 + r + 8*lh;
#pragma unroll
        for (int dt = 0; dt < 4; ++dt) {
            const int d = dt*16 + ln;
            O[((size_t)(bb*1024 + qn)) * 1024 + d*16 + hh] = (_Float16)(oacc[dt][r] * inv);
        }
    }
}

// ---------------------------------------------------------------------------
// Output projection: d_out = O(f16) @ Wo(f32) + bo -> f32.
// A tile staged by TDM (pure f16 copy); W tile converted cooperatively.
// ---------------------------------------------------------------------------
__global__ __launch_bounds__(256)
void out_proj_kernel(const _Float16* __restrict__ Xh, const float* __restrict__ W,
                     const float* __restrict__ bias, float* __restrict__ out)
{
    __shared__ _Float16 Ash[128 * 32];
    __shared__ _Float16 Bsh[128 * 32];

    const int tid  = threadIdx.x;
    const int lane = tid & 31;
    const int wave = tid >> 5;
    const int wr   = wave >> 2;
    const int wc   = wave & 3;
    const int m0   = blockIdx.x * 128;
    const int n0   = blockIdx.y * 128;
    const int lh   = lane >> 4;
    const int ln   = lane & 15;

    const unsigned int aoff = (unsigned int)(uintptr_t)&Ash[0];

    v8f acc[4][2];
#pragma unroll
    for (int i = 0; i < 4; ++i)
#pragma unroll
        for (int j = 0; j < 2; ++j) acc[i][j] = (v8f){0,0,0,0,0,0,0,0};

    for (int k0 = 0; k0 < PK; k0 += 32) {
        // TDM: A tile 128 rows x 32 f16, row stride 1024 f16 (wave 0 issues)
        if (wave == 0)
            tdm_load_2d(&Xh[(size_t)m0 * PK + k0], aoff, 1u, 32u, 128u, 1024ull);
        // W tile (overlapped with TDM): packed k-pairs -> b32 DS stores
        {
            const int nn = (tid & 15) * 8;
            const int kp = tid >> 4;
            const int k  = 2 * kp;
            const float4* r0 = (const float4*)&W[(size_t)(k0 + k)     * PN + n0 + nn];
            const float4* r1 = (const float4*)&W[(size_t)(k0 + k + 1) * PN + n0 + nn];
            float a[8], b[8];
            float4 t;
            t = r0[0]; a[0]=t.x; a[1]=t.y; a[2]=t.z; a[3]=t.w;
            t = r0[1]; a[4]=t.x; a[5]=t.y; a[6]=t.z; a[7]=t.w;
            t = r1[0]; b[0]=t.x; b[1]=t.y; b[2]=t.z; b[3]=t.w;
            t = r1[1]; b[4]=t.x; b[5]=t.y; b[6]=t.z; b[7]=t.w;
#pragma unroll
            for (int j = 0; j < 8; ++j)
                *(unsigned int*)&Bsh[(nn + j) * 32 + k] = pk2(a[j], b[j]);
        }
        if (wave == 0) __builtin_amdgcn_s_wait_tensorcnt(0);
        __syncthreads();

        const int kb = lh * 16;
        v16h bfrag[2];
#pragma unroll
        for (int nt = 0; nt < 2; ++nt)
            bfrag[nt] = *(const v16h*)&Bsh[(wc*32 + nt*16 + ln) * 32 + kb];
#pragma unroll
        for (int mt = 0; mt < 4; ++mt) {
            v16h a = *(const v16h*)&Ash[(wr*64 + mt*16 + ln) * 32 + kb];
#pragma unroll
            for (int nt = 0; nt < 2; ++nt)
                acc[mt][nt] = wmma_f16(a, bfrag[nt], acc[mt][nt]);
        }
        __syncthreads();
    }

#pragma unroll
    for (int mt = 0; mt < 4; ++mt)
#pragma unroll
        for (int nt = 0; nt < 2; ++nt)
#pragma unroll
            for (int r = 0; r < 8; ++r) {
                const int gm = m0 + wr*64 + mt*16 + r + 8*lh;
                const int gn = n0 + wc*32 + nt*16 + ln;
                out[(size_t)gm * PN + gn] = acc[mt][nt][r] + bias[gn];
            }
}

// ---------------------------------------------------------------------------
extern "C" void kernel_launch(void* const* d_in, const int* in_sizes, int n_in,
                              void* d_out, int out_size, void* d_ws, size_t ws_size,
                              hipStream_t stream)
{
    const float* q    = (const float*)d_in[0];
    const float* k    = (const float*)d_in[1];
    const float* v    = (const float*)d_in[2];
    const float* mask = (const float*)d_in[3];
    const float* Wq   = (const float*)d_in[4];
    const float* bq   = (const float*)d_in[5];
    const float* Wk   = (const float*)d_in[6];
    const float* bk   = (const float*)d_in[7];
    const float* Wv   = (const float*)d_in[8];
    const float* bv   = (const float*)d_in[9];
    const float* Wo   = (const float*)d_in[10];
    const float* bo   = (const float*)d_in[11];
    float* out        = (float*)d_out;

    const size_t QKV = (size_t)4 * 16 * 1024 * 64;
    _Float16* Qh = (_Float16*)d_ws;
    _Float16* Kh = Qh + QKV;
    _Float16* Vt = Kh + QKV;
    _Float16* O  = Vt + QKV;

    const float scale = 1.0f / 32.0f;   // 1024^-0.5

    dim3 g(32, 8);
    qkv_proj_kernel<<<g, 256, 0, stream>>>(q, Wq, bq, Qh, 0, scale);
    qkv_proj_kernel<<<g, 256, 0, stream>>>(k, Wk, bk, Kh, 0, 1.0f);
    qkv_proj_kernel<<<g, 256, 0, stream>>>(v, Wv, bv, Vt, 1, 1.0f);

    attn_kernel<<<dim3(16, 64), 128, 0, stream>>>(Qh, Kh, Vt, mask, O);

    out_proj_kernel<<<g, 256, 0, stream>>>(O, Wo, bo, out);
}
